// Transformer_10436770529689
// MI455X (gfx1250) — matrix operand order repaired and checked by reading, not compiled, not run on hardware
//
#include <hip/hip_runtime.h>

typedef __attribute__((ext_vector_type(8)))  float  v8f;
typedef __attribute__((ext_vector_type(8)))  __bf16 v8bf;
typedef __attribute__((ext_vector_type(16))) __bf16 v16bf;

#define NTOK   64
#define DIM    256
#define HEADS  8
#define HD     32
#define NWIN   4096
#define TOPK   51
#define QSCALE 0.17677669529663687f   // 32^-0.5

// ---- LDS layout (bytes) -------------------------------------------------
#define OFF_X    0        // bf16[64*256]  var / x / attn-out staging (32KB)
#define OFF_QK   32768    // bf16[64*512]  q(scaled)||k  (64KB); var_self f32[4096] overlaps
#define OFF_VT   98304    // bf16[256*64]  v transposed  (32KB)
#define OFF_MOD  131072   // f32 [4096]    mod matrix    (16KB)
#define OFF_ATT  147456   // bf16[8*64*64] attn per head (64KB); u64 rank keys overlap
#define OFF_REL  212992   // i32 [4096]    rel_index staged via async-to-LDS (16KB)
#define SMEM_BYTES 229376

// ---- WMMA fragment loaders (CDNA5 16-bit layouts, wave32) ---------------
// A (16x32): lane l -> row m0+(l&15); elems 0..7 = K k0+8*(l>>4)+i (16B chunk),
// elems 8..15 = K k0+16+8*(l>>4)+i (second 16B chunk).
__device__ inline v16bf load_a(const __bf16* p, int stride, int m0, int k0, int lane) {
  const __bf16* row = p + (m0 + (lane & 15)) * stride + k0 + ((lane >> 4) << 3);
  union { v16bf v; v8bf h[2]; } u;
  u.h[0] = *(const v8bf*)(row);
  u.h[1] = *(const v8bf*)(row + 16);
  return u.v;
}
// B (32x16): lane l -> col n0+(l&15); elems 0..7 = K k0+8*(l>>4)+i,
// elems 8..15 = K k0+16+8*(l>>4)+i, from "n-major" storage p[n*stride + k].
__device__ inline v16bf load_b(const __bf16* p, int stride, int n0, int k0, int lane) {
  const __bf16* row = p + (n0 + (lane & 15)) * stride + k0 + ((lane >> 4) << 3);
  union { v16bf v; v8bf h[2]; } u;
  u.h[0] = *(const v8bf*)(row);
  u.h[1] = *(const v8bf*)(row + 16);
  return u.v;
}
__device__ inline v8f wmma_bf16(v16bf a, v16bf b, v8f c) {
  return __builtin_amdgcn_wmma_f32_16x16x32_bf16(false, a, false, b, (short)0, c, false, false);
}

// ---- weight prep: fp32 -> bf16, transposed to n-major -------------------
__global__ void prep_weights(const float* __restrict__ Wqkv,
                             const float* __restrict__ Wproj,
                             __bf16* __restrict__ wqkv_t,
                             __bf16* __restrict__ wproj_t) {
  int n = blockIdx.x;        // 0..767
  int k = threadIdx.x;       // 0..255
  wqkv_t[n * DIM + k] = (__bf16)Wqkv[k * (3 * DIM) + n];
  if (n < DIM) wproj_t[n * DIM + k] = (__bf16)Wproj[k * DIM + n];
}

// ---- fused windowed attention ------------------------------------------
__global__ __launch_bounds__(256) void swin_attn_kernel(
    const float* __restrict__ x, const float* __restrict__ var,
    const float* __restrict__ b_qkv, const float* __restrict__ b_proj,
    const float* __restrict__ bias_table, const int* __restrict__ rel_index,
    const __bf16* __restrict__ wqkv_t, const __bf16* __restrict__ wproj_t,
    float* __restrict__ out) {
  extern __shared__ char smem[];
  __bf16* xb   = (__bf16*)(smem + OFF_X);
  __bf16* qk   = (__bf16*)(smem + OFF_QK);
  float*  vsf  = (float*)(smem + OFF_QK);
  __bf16* vT   = (__bf16*)(smem + OFF_VT);
  float*  modf = (float*)(smem + OFF_MOD);
  __bf16* att  = (__bf16*)(smem + OFF_ATT);
  unsigned long long* keys = (unsigned long long*)(smem + OFF_ATT);
  int*    relL = (int*)(smem + OFF_REL);

  const int tid  = threadIdx.x;
  const int wave = tid >> 5;
  const int lane = tid & 31;
  const int half = lane >> 4;
  const int l16  = lane & 15;
  const size_t base = (size_t)blockIdx.x * NTOK * DIM;

  // Stage 0: kick off async copy of rel_index (16KB) into LDS; it overlaps
  // stages 1-5 and is waited on (ASYNCcnt) just before the attention stage.
  for (int i = tid; i < NTOK * NTOK / 4; i += 256) {   // 1024 x 16B
    unsigned ldsa = (unsigned)(uintptr_t)(relL + i * 4);
    unsigned long long ga = (unsigned long long)(uintptr_t)(rel_index + i * 4);
    asm volatile("global_load_async_to_lds_b128 %0, %1, off"
                 :: "v"(ldsa), "v"(ga) : "memory");
  }

  // Stage 1: var -> bf16 LDS
  {
    const float4* src = (const float4*)(var + base);
    for (int i = tid; i < NTOK * DIM / 4; i += 256) {
      float4 f = src[i];
      __bf16* d = xb + i * 4;
      d[0] = (__bf16)f.x; d[1] = (__bf16)f.y; d[2] = (__bf16)f.z; d[3] = (__bf16)f.w;
    }
  }
  __syncthreads();

  // Stage 2: var_self = var . var^T  (64x64, K=256) -> vsf (f32)
  for (int t = 0; t < 2; ++t) {
    int tile = wave * 2 + t;
    int tm = tile >> 2, tn = tile & 3;
    v8f acc = {};
    for (int ks = 0; ks < 8; ++ks) {
      v16bf a  = load_a(xb, DIM, tm * 16, ks * 32, lane);
      v16bf bb = load_b(xb, DIM, tn * 16, ks * 32, lane);  // B[k][n] = var[n][k]
      acc = wmma_bf16(a, bb, acc);
    }
    for (int r = 0; r < 8; ++r)
      vsf[(tm * 16 + r + half * 8) * 64 + tn * 16 + l16] = acc[r];
  }
  __syncthreads();

  // Stage 3a: build sortable 64-bit keys: monotonic uint of f32, tie-broken
  // by smaller index first (matches jax.lax.top_k ordering).
  for (int idx = tid; idx < NTOK * NTOK; idx += 256) {
    unsigned u = __float_as_uint(vsf[idx]);
    u = (u & 0x80000000u) ? ~u : (u | 0x80000000u);
    keys[idx] = ((unsigned long long)u << 6) | (unsigned)(63 - (idx & 63));
  }
  __syncthreads();
  // Stage 3b: rank scan -> mod;  Stage 4: x -> bf16 LDS
  for (int idx = tid; idx < NTOK * NTOK; idx += 256) {
    const unsigned long long* krow = keys + (idx >> 6) * 64;
    unsigned long long cm = keys[idx];
    int rank = 0;
#pragma unroll 8
    for (int j = 0; j < 64; ++j) rank += (krow[j] > cm);
    modf[idx] = (rank < TOPK) ? 1.0f : 1.2f;
  }
  {
    const float4* src = (const float4*)(x + base);
    for (int i = tid; i < NTOK * DIM / 4; i += 256) {
      float4 f = src[i];
      __bf16* d = xb + i * 4;
      d[0] = (__bf16)f.x; d[1] = (__bf16)f.y; d[2] = (__bf16)f.z; d[3] = (__bf16)f.w;
    }
  }
  __syncthreads();

  // Stage 5: qkv = x @ Wqkv + b  (64x768, K=256); q pre-scaled, v transposed
  for (int t = 0; t < 24; ++t) {
    int tile = wave * 24 + t;
    int tm = tile & 3, tn = tile >> 2;      // tn in 0..47
    int n0 = tn * 16;
    v8f acc = {};
    for (int ks = 0; ks < 8; ++ks) {
      v16bf a  = load_a(xb, DIM, tm * 16, ks * 32, lane);
      v16bf bb = load_b(wqkv_t, DIM, n0, ks * 32, lane);   // global bf16, n-major
      acc = wmma_bf16(a, bb, acc);
    }
    int col = n0 + l16;
    float bias = b_qkv[col];
    int seg = __builtin_amdgcn_readfirstlane(n0 >> 8);     // wave-uniform: 0=q 1=k 2=v
    if (seg == 0) {
      for (int r = 0; r < 8; ++r)
        qk[(tm * 16 + r + half * 8) * 512 + col] = (__bf16)((acc[r] + bias) * QSCALE);
    } else if (seg == 1) {
      for (int r = 0; r < 8; ++r)
        qk[(tm * 16 + r + half * 8) * 512 + col] = (__bf16)(acc[r] + bias);
    } else {
      for (int r = 0; r < 8; ++r)
        vT[(col - 2 * DIM) * 64 + tm * 16 + r + half * 8] = (__bf16)(acc[r] + bias);
    }
  }
  asm volatile("s_wait_asynccnt 0" ::: "memory");  // rel_index now resident in LDS
  __syncthreads();

  // Stage 6: one wave per head
  {
    const int h = wave;
    v8f dots[4][4];
    for (int tm = 0; tm < 4; ++tm) {
      v16bf a = load_a(qk, 512, tm * 16, h * HD, lane);          // q rows, K=32
      for (int tn = 0; tn < 4; ++tn) {
        v16bf bb = load_b(qk + DIM, 512, tn * 16, h * HD, lane); // B[k][n]=k[n][k]
        v8f z = {};
        dots[tm][tn] = wmma_bf16(a, bb, z);
      }
    }
    // mod * bias + softmax (row M lives across a 16-lane half)
    for (int tm = 0; tm < 4; ++tm) {
      for (int r = 0; r < 8; ++r) {
        int m = tm * 16 + r + half * 8;
        for (int tn = 0; tn < 4; ++tn) {
          int n = tn * 16 + l16;
          float d = dots[tm][tn][r] * modf[m * 64 + n] +
                    bias_table[relL[m * 64 + n] * HEADS + h];
          dots[tm][tn][r] = d;
        }
        float mx = -1e30f;
        for (int tn = 0; tn < 4; ++tn) mx = fmaxf(mx, dots[tm][tn][r]);
        for (int o = 1; o < 16; o <<= 1) mx = fmaxf(mx, __shfl_xor(mx, o, 32));
        float s = 0.f;
        for (int tn = 0; tn < 4; ++tn) {
          float e = __expf(dots[tm][tn][r] - mx);
          dots[tm][tn][r] = e; s += e;
        }
        for (int o = 1; o < 16; o <<= 1) s += __shfl_xor(s, o, 32);
        float inv = 1.0f / s;
        for (int tn = 0; tn < 4; ++tn)
          att[h * 4096 + m * 64 + tn * 16 + l16] = (__bf16)(dots[tm][tn][r] * inv);
      }
    }
    __syncthreads();
    // out_h = attn @ v_h  (64x32, K=64)
    const __bf16* attH = att + h * 4096;
    const __bf16* vTH  = vT + h * HD * 64;
    for (int tm = 0; tm < 4; ++tm) {
      for (int tn = 0; tn < 2; ++tn) {
        v8f o = {};
        for (int ks = 0; ks < 2; ++ks) {
          v16bf a  = load_a(attH, 64, tm * 16, ks * 32, lane);
          v16bf bb = load_b(vTH, 64, tn * 16, ks * 32, lane);   // B[k][n]=vT[n][k]
          o = wmma_bf16(a, bb, o);
        }
        for (int r = 0; r < 8; ++r)
          xb[(tm * 16 + r + half * 8) * DIM + h * HD + tn * 16 + l16] = (__bf16)o[r];
      }
    }
  }
  __syncthreads();

  // Stage 7: out = attn_out @ Wproj + b  (64x256, K=256)
  for (int t = 0; t < 8; ++t) {
    int tm = t & 3;
    int tn = wave * 2 + (t >> 2);
    v8f acc = {};
    for (int ks = 0; ks < 8; ++ks) {
      v16bf a  = load_a(xb, DIM, tm * 16, ks * 32, lane);
      v16bf bb = load_b(wproj_t, DIM, tn * 16, ks * 32, lane);
      acc = wmma_bf16(a, bb, acc);
    }
    int col = tn * 16 + l16;
    float bp = b_proj[col];
    for (int r = 0; r < 8; ++r)
      out[base + (size_t)(tm * 16 + r + half * 8) * DIM + col] = acc[r] + bp;
  }
}

extern "C" void kernel_launch(void* const* d_in, const int* in_sizes, int n_in,
                              void* d_out, int out_size, void* d_ws, size_t ws_size,
                              hipStream_t stream) {
  const float* x          = (const float*)d_in[0];
  const float* var        = (const float*)d_in[1];
  const float* W_qkv      = (const float*)d_in[2];
  const float* b_qkv      = (const float*)d_in[3];
  const float* W_proj     = (const float*)d_in[4];
  const float* b_proj     = (const float*)d_in[5];
  const float* bias_table = (const float*)d_in[6];
  const int*   rel_index  = (const int*)d_in[7];
  float* out = (float*)d_out;

  __bf16* wqkv_t  = (__bf16*)d_ws;                                   // 768*256 bf16
  __bf16* wproj_t = (__bf16*)((char*)d_ws + (size_t)768 * DIM * 2);  // 256*256 bf16

  (void)hipFuncSetAttribute((const void*)swin_attn_kernel,
                            hipFuncAttributeMaxDynamicSharedMemorySize, SMEM_BYTES);

  prep_weights<<<768, 256, 0, stream>>>(W_qkv, W_proj, wqkv_t, wproj_t);
  swin_attn_kernel<<<NWIN, 256, SMEM_BYTES, stream>>>(
      x, var, b_qkv, b_proj, bias_table, rel_index, wqkv_t, wproj_t, out);
}
